// LearnableLSHAttention_10926396801633
// MI455X (gfx1250) — matrix operand order repaired and checked by reading, not compiled, not run on hardware
//
#include <hip/hip_runtime.h>

#define NTOK   2048   // B*S
#define DMODEL 1024
#define NHEAD  8
#define HDIM   128
#define NBUCK  64
#define NEGINF (-1.0e30f)

#define KLD     136               // padded LDS row (elements) for K tile: 16B-chunk bank spread
#define KTILE_E (64 * KLD)        // elements per K-tile buffer

typedef __attribute__((ext_vector_type(16))) __bf16 v16bf;
typedef __attribute__((ext_vector_type(8)))  float  v8f;

static __device__ __forceinline__ v8f wmma_bf16(v16bf a, v16bf b, v8f c) {
    // v_wmma_f32_16x16x32_bf16: D = A(16x32) * B(32x16) + C(16x16 f32)
    return __builtin_amdgcn_wmma_f32_16x16x32_bf16(false, a, false, b, (short)0, c,
                                                   false, false);
}

// ---- CDNA5 async global->LDS (ASYNCcnt) -------------------------------------
// LDS byte address comes from the vdst VGPR; generic addr low 32 bits of a __shared__ ptr
// are the LDS byte offset (flat-LDS aperture truncation).
static __device__ __forceinline__ void async_ld_b128(unsigned int lds_off,
                                                     const void* gaddr) {
    asm volatile("global_load_async_to_lds_b128 %0, %1, off"
                 :: "v"(lds_off), "v"((unsigned long long)(size_t)gaddr)
                 : "memory");
}
static __device__ __forceinline__ void wait_async_le4() {
    asm volatile("s_wait_asynccnt 0x4" ::: "memory");
}
static __device__ __forceinline__ void wait_async_0() {
    asm volatile("s_wait_asynccnt 0x0" ::: "memory");
}

// A-fragment (16x32 bf16, MxK). Per ISA layout:
//  lanes 0-15  (M=lane):    elems 0..7 -> K = k0+0..7,  elems 8..15 -> K = k0+16..23
//  lanes 16-31 (M=lane-16): elems 0..7 -> K = k0+8..15, elems 8..15 -> K = k0+24..31
static __device__ __forceinline__ v16bf load_frag_a(const __bf16* base, int ld,
                                                    int m0, int k0, int lane) {
    const int row = m0 + (lane & 15);
    const int kb  = k0 + ((lane >> 4) << 3);
    const __bf16* p = base + (size_t)row * ld + kb;
    v16bf a;
    ((uint4*)&a)[0] = *(const uint4*)(p);
    ((uint4*)&a)[1] = *(const uint4*)(p + 16);
    return a;
}

// B-fragment (32x16 bf16, KxN), source stored transposed: base[col][k] contiguous in k.
static __device__ __forceinline__ v16bf load_frag_b(const __bf16* base, int ld,
                                                    int n0, int k0, int lane) {
    const int col = n0 + (lane & 15);
    const int kb  = k0 + ((lane >> 4) << 3);
    const __bf16* p = base + (size_t)col * ld + kb;
    v16bf b;
    ((uint4*)&b)[0] = *(const uint4*)(p);
    ((uint4*)&b)[1] = *(const uint4*)(p + 16);
    return b;
}

// ---------------------------------------------------------------- converts
__global__ void __launch_bounds__(256) k_f32_to_bf16(const float* __restrict__ in,
                                                     __bf16* __restrict__ out, int n) {
    int i = blockIdx.x * 256 + threadIdx.x;
    if (i < n) out[i] = (__bf16)in[i];
}

// wt[c][k] = W[k][c]  (bf16)
__global__ void __launch_bounds__(256) k_f32_to_bf16_T(const float* __restrict__ in,
                                                       __bf16* __restrict__ out,
                                                       int rows, int cols) {
    int i = blockIdx.x * 256 + threadIdx.x;
    if (i < rows * cols) {
        int k = i / cols, c = i % cols;
        out[(size_t)c * rows + k] = (__bf16)in[i];
    }
}

// ---------------------------------------------------------------- GEMM
// C[m][n] = sum_k A[m][k] * Bt[n][k] + bias[n]
// mode 0: out bf16 [M][N] ; mode 1: out bf16 [N][M] (transposed) ; mode 2: out f32 [M][N]
__global__ void __launch_bounds__(128)
k_gemm_bf16(const __bf16* __restrict__ A, const __bf16* __restrict__ Bt,
            const float* __restrict__ bias, void* __restrict__ Out,
            int M, int N, int K, int mode) {
    const int wave = threadIdx.x >> 5, lane = threadIdx.x & 31;
    const int m0 = blockIdx.y * 64 + (wave >> 1) * 32;
    const int n0 = blockIdx.x * 64 + (wave & 1) * 32;

    v8f c00 = {}, c01 = {}, c10 = {}, c11 = {};
    for (int k0 = 0; k0 < K; k0 += 32) {
        v16bf a0 = load_frag_a(A, K, m0,      k0, lane);
        v16bf a1 = load_frag_a(A, K, m0 + 16, k0, lane);
        v16bf b0 = load_frag_b(Bt, K, n0,      k0, lane);
        v16bf b1 = load_frag_b(Bt, K, n0 + 16, k0, lane);
        c00 = wmma_bf16(a0, b0, c00);
        c01 = wmma_bf16(a0, b1, c01);
        c10 = wmma_bf16(a1, b0, c10);
        c11 = wmma_bf16(a1, b1, c11);
    }

    v8f cf[2][2] = {{c00, c01}, {c10, c11}};
    for (int mi = 0; mi < 2; ++mi)
        for (int ni = 0; ni < 2; ++ni) {
            const int col  = n0 + ni * 16 + (lane & 15);
            const int rowb = m0 + mi * 16 + ((lane >> 4) << 3);
            const float bsv = bias[col];
            if (mode == 0) {
                __bf16* o = (__bf16*)Out;
                for (int r = 0; r < 8; ++r)
                    o[(size_t)(rowb + r) * N + col] = (__bf16)(cf[mi][ni][r] + bsv);
            } else if (mode == 1) {
                __bf16* o = (__bf16*)Out;   // [N][M]
                __bf16 tmp[8];
                for (int r = 0; r < 8; ++r) tmp[r] = (__bf16)(cf[mi][ni][r] + bsv);
                *(uint4*)(o + (size_t)col * M + rowb) = *(const uint4*)tmp;
            } else {
                float* o = (float*)Out;
                for (int r = 0; r < 8; ++r)
                    o[(size_t)(rowb + r) * N + col] = cf[mi][ni][r] + bsv;
            }
        }
}

// ---------------------------------------------------------------- LSH bucketing
__global__ void __launch_bounds__(64)
k_hash_argmax(const __bf16* __restrict__ qb, const float* __restrict__ hp,
              int* __restrict__ buckets) {
    const int id = blockIdx.x;
    const int h = id / NTOK, n = id % NTOK;
    const int f = threadIdx.x;
    const __bf16* q = qb + (size_t)n * DMODEL + h * HDIM;
    const float*  w = hp + (size_t)h * HDIM * NBUCK + f;
    float s = 0.f;
    for (int d = 0; d < HDIM; ++d) s += (float)q[d] * w[(size_t)d * NBUCK];
    __shared__ float sv[64];
    __shared__ int   si[64];
    sv[f] = s; si[f] = f;
    __syncthreads();
    for (int off = 32; off > 0; off >>= 1) {
        if (f < off) {
            if (sv[f + off] > sv[f] || (sv[f + off] == sv[f] && si[f + off] < si[f])) {
                sv[f] = sv[f + off]; si[f] = si[f + off];
            }
        }
        __syncthreads();
    }
    if (f == 0) buckets[h * NTOK + n] = si[0];
}

// ---------------------------------------------------------------- masked flash attention
// grid (NTOK/64, NHEAD), 256 threads (8 waves). Block = 64 query rows of one head.
// K tiles are staged into LDS with double-buffered async copies (ASYNCcnt-tracked);
// each wave issues 4 async b128 per tile -> s_wait_asynccnt 4 drains the current
// tile while the next tile stays in flight (async loads complete in order).
__global__ void __launch_bounds__(256)
k_lsh_attn(const __bf16* __restrict__ qb, const __bf16* __restrict__ kb,
           const __bf16* __restrict__ vt, const int* __restrict__ buckets,
           __bf16* __restrict__ ob) {
    const int h   = blockIdx.y;
    const int m0g = blockIdx.x * 64;
    const int tid = threadIdx.x;
    const int wave = tid >> 5, lane = tid & 31;
    const float SCALE = 0.08838834764831845f;   // 1/sqrt(128)

    __shared__ alignas(16) float  s_s[64 * 64];
    __shared__ alignas(16) __bf16 p_s[64 * 64];
    __shared__ alignas(16) __bf16 ks_s[2 * KTILE_E];   // double-buffered K tile
    __shared__ float red_s[256];
    __shared__ int   bq_s[64], bk_s[64];
    __shared__ float mrow[64], lrow[64], arow[64];

    if (tid < 64) {
        bq_s[tid] = buckets[h * NTOK + m0g + tid];
        mrow[tid] = NEGINF;
        lrow[tid] = 0.f;
    }

    const __bf16* Qh = qb + h * HDIM;                    // [NTOK][DMODEL] view
    const __bf16* Kh = kb + h * HDIM;
    const __bf16* Vh = vt + (size_t)h * HDIM * NTOK;     // [HDIM][NTOK]

    // prologue: async-stage K tile 0 into buffer 0 (4 x b128 per thread)
    for (int c = 0; c < 4; ++c) {
        const int chunk = tid + c * 256;                 // 0..1023 (16B chunks)
        const int krow  = chunk >> 4;                    // 16 chunks (256B) per row
        const int c16   = chunk & 15;
        const unsigned int loff =
            (unsigned int)(size_t)(&ks_s[krow * KLD]) + c16 * 16;
        async_ld_b128(loff, (const char*)(Kh + (size_t)krow * DMODEL) + c16 * 16);
    }

    v8f of[4] = {};   // O accumulator: cols 16*wave..+15, rows 64

    for (int n0 = 0; n0 < NTOK; n0 += 64) {
        const int cur = (n0 >> 6) & 1;
        __syncthreads();   // prior iteration fully done (incl. reads of buf cur^1)
        if (tid < 64) bk_s[tid] = buckets[h * NTOK + n0 + tid];

        if (n0 + 64 < NTOK) {
            // issue next K tile into the other buffer, then drain current tile
            const int nxt = cur ^ 1;
            for (int c = 0; c < 4; ++c) {
                const int chunk = tid + c * 256;
                const int krow  = chunk >> 4;
                const int c16   = chunk & 15;
                const unsigned int loff =
                    (unsigned int)(size_t)(&ks_s[nxt * KTILE_E + krow * KLD]) + c16 * 16;
                async_ld_b128(loff,
                    (const char*)(Kh + (size_t)(n0 + 64 + krow) * DMODEL) + c16 * 16);
            }
            wait_async_le4();
            // prefetch next V tile (this wave's 16 value-dim rows)
            __builtin_prefetch(Vh + (size_t)(wave * 16 + (lane & 15)) * NTOK + n0 + 64,
                               0, 3);
        } else {
            wait_async_0();
        }
        __syncthreads();   // whole K tile visible to all waves; bk_s ready

        // ---- S = Q K^T (2 frags per wave); K from LDS tile ----
        const __bf16* Kt = ks_s + cur * KTILE_E;
        const int ni = wave & 3;
        const int mih = wave >> 2;
        v8f sf0 = {}, sf1 = {};
        for (int k0 = 0; k0 < HDIM; k0 += 32) {
            v16bf bf = load_frag_b(Kt, KLD, ni * 16, k0, lane);
            v16bf a0 = load_frag_a(Qh, DMODEL, m0g + (mih * 2 + 0) * 16, k0, lane);
            v16bf a1 = load_frag_a(Qh, DMODEL, m0g + (mih * 2 + 1) * 16, k0, lane);
            sf0 = wmma_bf16(a0, bf, sf0);
            sf1 = wmma_bf16(a1, bf, sf1);
        }
        {   // masked + scaled write to LDS
            const int col = ni * 16 + (lane & 15);
            const int bkv = bk_s[col];
            for (int t = 0; t < 2; ++t) {
                v8f s = t ? sf1 : sf0;
                const int rowb = (mih * 2 + t) * 16 + ((lane >> 4) << 3);
                for (int r = 0; r < 8; ++r) {
                    const int row = rowb + r;
                    s_s[row * 64 + col] = (bq_s[row] == bkv) ? s[r] * SCALE : NEGINF;
                }
            }
        }
        __syncthreads();

        // ---- online softmax: 4 threads per row, 16 cols each ----
        const int row = tid >> 2;
        const int cb  = (tid & 3) * 16;
        float pm = NEGINF;
        for (int j = 0; j < 16; ++j) pm = fmaxf(pm, s_s[row * 64 + cb + j]);
        red_s[tid] = pm;
        __syncthreads();
        const float rowmax = fmaxf(fmaxf(red_s[row * 4 + 0], red_s[row * 4 + 1]),
                                   fmaxf(red_s[row * 4 + 2], red_s[row * 4 + 3]));
        const float mold = mrow[row];
        const float newm = fmaxf(mold, rowmax);
        __syncthreads();
        float psum = 0.f;
        for (int j = 0; j < 16; ++j) {
            const float sv = s_s[row * 64 + cb + j];
            const float p = (sv < 0.5f * NEGINF) ? 0.f : __expf(sv - newm);
            p_s[row * 64 + cb + j] = (__bf16)p;
            psum += p;
        }
        red_s[tid] = psum;
        __syncthreads();
        if ((tid & 3) == 0) {
            const float rsum = red_s[tid] + red_s[tid + 1] + red_s[tid + 2] + red_s[tid + 3];
            const float alpha = __expf(mold - newm);
            arow[row] = alpha;
            lrow[row] = lrow[row] * alpha + rsum;
            mrow[row] = newm;
        }
        __syncthreads();

        // ---- O = O*alpha + P @ V ----
        for (int mi = 0; mi < 4; ++mi) {
            const int rowb = mi * 16 + ((lane >> 4) << 3);
            for (int r = 0; r < 8; ++r) of[mi][r] *= arow[rowb + r];
        }
        for (int kk = 0; kk < 64; kk += 32) {
            v16bf bf = load_frag_b(Vh, NTOK, wave * 16, n0 + kk, lane);
            for (int mi = 0; mi < 4; ++mi) {
                v16bf af = load_frag_a(p_s, 64, mi * 16, kk, lane);
                of[mi] = wmma_bf16(af, bf, of[mi]);
            }
        }
    }

    __syncthreads();
    const int colg = h * HDIM + wave * 16 + (lane & 15);
    for (int mi = 0; mi < 4; ++mi) {
        const int rowb = mi * 16 + ((lane >> 4) << 3);
        for (int r = 0; r < 8; ++r) {
            const int row = rowb + r;
            ob[(size_t)(m0g + row) * DMODEL + colg] = (__bf16)(of[mi][r] / lrow[row]);
        }
    }
}

// ---------------------------------------------------------------- launch
extern "C" void kernel_launch(void* const* d_in, const int* in_sizes, int n_in,
                              void* d_out, int out_size, void* d_ws, size_t ws_size,
                              hipStream_t stream) {
    const float* x  = (const float*)d_in[0];
    const float* Wq = (const float*)d_in[1];
    const float* bq = (const float*)d_in[2];
    const float* Wk = (const float*)d_in[3];
    const float* bk = (const float*)d_in[4];
    const float* Wv = (const float*)d_in[5];
    const float* bv = (const float*)d_in[6];
    const float* Wo = (const float*)d_in[7];
    const float* bo = (const float*)d_in[8];
    const float* hp = (const float*)d_in[9];

    char* ws = (char*)d_ws;
    const size_t MB = 1024 * 1024;
    __bf16* xb  = (__bf16*)(ws + 0);        // [NTOK][D]      4 MB
    __bf16* wqT = (__bf16*)(ws + 4 * MB);   // [D][D]         2 MB
    __bf16* wkT = (__bf16*)(ws + 6 * MB);
    __bf16* wvT = (__bf16*)(ws + 8 * MB);
    __bf16* woT = (__bf16*)(ws + 10 * MB);
    __bf16* qbf = (__bf16*)(ws + 12 * MB);  // [NTOK][D]      4 MB
    __bf16* kbf = (__bf16*)(ws + 16 * MB);
    __bf16* vtb = (__bf16*)(ws + 20 * MB);  // [D][NTOK]      4 MB
    __bf16* obf = (__bf16*)(ws + 24 * MB);  // [NTOK][D]      4 MB
    int*    bkt = (int*)   (ws + 28 * MB);  // [H][NTOK]      64 KB

    const int NX = NTOK * DMODEL;           // 2M
    const int NW = DMODEL * DMODEL;         // 1M

    k_f32_to_bf16<<<(NX + 255) / 256, 256, 0, stream>>>(x, xb, NX);
    k_f32_to_bf16_T<<<(NW + 255) / 256, 256, 0, stream>>>(Wq, wqT, DMODEL, DMODEL);
    k_f32_to_bf16_T<<<(NW + 255) / 256, 256, 0, stream>>>(Wk, wkT, DMODEL, DMODEL);
    k_f32_to_bf16_T<<<(NW + 255) / 256, 256, 0, stream>>>(Wv, wvT, DMODEL, DMODEL);
    k_f32_to_bf16_T<<<(NW + 255) / 256, 256, 0, stream>>>(Wo, woT, DMODEL, DMODEL);

    dim3 gg(DMODEL / 64, NTOK / 64);        // (16, 32)
    k_gemm_bf16<<<gg, 128, 0, stream>>>(xb, wqT, bq, qbf, NTOK, DMODEL, DMODEL, 0);
    k_gemm_bf16<<<gg, 128, 0, stream>>>(xb, wkT, bk, kbf, NTOK, DMODEL, DMODEL, 0);
    k_gemm_bf16<<<gg, 128, 0, stream>>>(xb, wvT, bv, vtb, NTOK, DMODEL, DMODEL, 1);

    k_hash_argmax<<<NHEAD * NTOK, 64, 0, stream>>>(qbf, hp, bkt);

    k_lsh_attn<<<dim3(NTOK / 64, NHEAD), 256, 0, stream>>>(qbf, kbf, vtb, bkt, obf);

    k_gemm_bf16<<<gg, 128, 0, stream>>>(obf, woT, bo, (float*)d_out,
                                        NTOK, DMODEL, DMODEL, 2);
}
